// GlobalContext_16303695855876
// MI455X (gfx1250) — matrix operand order repaired and checked by reading, not compiled, not run on hardware
//
#include <hip/hip_runtime.h>

#define NROWS 8192
#define CDIM  256

typedef __attribute__((ext_vector_type(16))) _Float16 v16h;
typedef __attribute__((ext_vector_type(8)))  _Float16 v8h;
typedef __attribute__((ext_vector_type(8)))  float    v8f;
typedef __attribute__((ext_vector_type(4)))  float    f32x4;

// ---------------------------------------------------------------------------
// kT: wrt[k*C + c] = Wr[c*C + k]  (transpose so kA reads broadcast rows)
// ---------------------------------------------------------------------------
__global__ void kT(const float* __restrict__ wr, float* __restrict__ wrt) {
  int k = blockIdx.x;
  int c = threadIdx.x;
  wrt[k * CDIM + c] = wr[c * CDIM + k];
}

// ---------------------------------------------------------------------------
// kA: r = x @ Wr^T in f32; also rhT[c*N + i] = (f16) r[i][c]  (transposed f16
// copy so the WMMA B fragments in kMain are contiguous b128 loads).
// Block = 256 threads, 4 rows of x per block staged in LDS.
// ---------------------------------------------------------------------------
__global__ void kA(const float* __restrict__ x, const float* __restrict__ wrt,
                   float* __restrict__ r, _Float16* __restrict__ rhT) {
  __shared__ float xl[4][CDIM];
  const int i0 = blockIdx.x * 4;
  const int c  = threadIdx.x;
  #pragma unroll
  for (int m = 0; m < 4; ++m) xl[m][c] = x[(size_t)(i0 + m) * CDIM + c];
  __syncthreads();

  float a0 = 0.f, a1 = 0.f, a2 = 0.f, a3 = 0.f;
  for (int k = 0; k < CDIM; ++k) {
    float w = wrt[k * CDIM + c];   // coalesced across lanes
    a0 += xl[0][k] * w;            // LDS broadcast reads
    a1 += xl[1][k] * w;
    a2 += xl[2][k] * w;
    a3 += xl[3][k] * w;
  }
  r[(size_t)(i0 + 0) * CDIM + c] = a0;
  r[(size_t)(i0 + 1) * CDIM + c] = a1;
  r[(size_t)(i0 + 2) * CDIM + c] = a2;
  r[(size_t)(i0 + 3) * CDIM + c] = a3;
  rhT[(size_t)c * NROWS + i0 + 0] = (_Float16)a0;
  rhT[(size_t)c * NROWS + i0 + 1] = (_Float16)a1;
  rhT[(size_t)c * NROWS + i0 + 2] = (_Float16)a2;
  rhT[(size_t)c * NROWS + i0 + 3] = (_Float16)a3;
}

// ---------------------------------------------------------------------------
// kB: s[i] = r[i,:] . wc   (blocks 0..N-1);  s[N] = c0 = br.wc + bc (block N)
// ---------------------------------------------------------------------------
__global__ void kB(const float* __restrict__ r, const float* __restrict__ wc,
                   const float* __restrict__ br, const float* __restrict__ bc,
                   float* __restrict__ s) {
  __shared__ float red[CDIM];
  const int i = blockIdx.x;
  const int t = threadIdx.x;
  float v = (i < NROWS) ? r[(size_t)i * CDIM + t] * wc[t] : br[t] * wc[t];
  red[t] = v;
  __syncthreads();
  #pragma unroll
  for (int o = CDIM / 2; o > 0; o >>= 1) {
    if (t < o) red[t] += red[t + o];
    __syncthreads();
  }
  if (t == 0) s[i] = (i < NROWS) ? red[0] : red[0] + bc[0];
}

// ---------------------------------------------------------------------------
// kMain: per wave, 16 rows x all 256 cols.
//   out[i,c] = (r[i,c] + br[c]) * m_i + (Z @ r)[i,c] / N,  m_i = mean_j Z_ij
// Z tiles (16x32 f16) are generated in-register in the WMMA A layout and fed
// straight into v_wmma_f32_16x16x32_f16 against B tiles from rhT.
// ---------------------------------------------------------------------------
__global__ void __launch_bounds__(128)
kMain(const float* __restrict__ s,        // [N+1], s[N] = c0
      const _Float16* __restrict__ rhT,   // [C][N]
      const float* __restrict__ r,        // [N][C]
      const float* __restrict__ br,       // [C]
      float* __restrict__ out) {          // [N][C]
  const int lane  = threadIdx.x & 31;
  const int wave  = threadIdx.x >> 5;
  const int i0    = blockIdx.x * 64 + wave * 16;
  const int Mrow  = lane & 15;            // A-matrix row / C-matrix column idx
  const int khalf = lane >> 4;            // lane-half selects K group
  const float c0  = s[NROWS];
  const float si  = s[i0 + Mrow] + c0;    // folded per-row bias

  const v8f vzero = {0.f, 0.f, 0.f, 0.f, 0.f, 0.f, 0.f, 0.f};
  v8f acc[16];
  #pragma unroll
  for (int t = 0; t < 16; ++t) acc[t] = vzero;

  float msum = 0.0f;

  for (int j0 = 0; j0 < NROWS; j0 += 32) {
    // s values for this lane's 16 A-fragment elements.
    // A 16-bit 16x32 layout: element e -> K = (e<8?0:16) + khalf*8 + (e&7)
    const float* sj = s + j0 + khalf * 8;
    f32x4 sA0 = *(const f32x4*)(sj);        // K = khalf*8 + 0..3   (e 0..3)
    f32x4 sA1 = *(const f32x4*)(sj + 4);    // K = khalf*8 + 4..7   (e 4..7)
    f32x4 sB0 = *(const f32x4*)(sj + 16);   // K = 16+khalf*8+0..3  (e 8..11)
    f32x4 sB1 = *(const f32x4*)(sj + 20);   // K = 16+khalf*8+4..7  (e 12..15)
    float sv[16];
    #pragma unroll
    for (int e = 0; e < 4; ++e) {
      sv[e]      = sA0[e];
      sv[e + 4]  = sA1[e];
      sv[e + 8]  = sB0[e];
      sv[e + 12] = sB1[e];
    }

    // Build Z tile (A fragment) + accumulate row sums for m_i.
    v16h af;
    #pragma unroll
    for (int e = 0; e < 16; ++e) {
      float arg = si + sv[e];
      float z   = __builtin_amdgcn_rcpf(1.0f + __expf(-arg));  // sigmoid
      msum += z;
      af[e] = (_Float16)z;
    }

    // 16 column tiles of B = r[j0..j0+31, :] from the transposed f16 copy.
    // B 16-bit 32x16 layout: lane -> N = lane&15, element e -> K = (e<8?0:16) + khalf*8 + (e&7)
    const _Float16* bbase = rhT + (j0 + khalf * 8);
    #pragma unroll
    for (int t = 0; t < 16; ++t) {
      const _Float16* bp = bbase + (size_t)(t * 16 + Mrow) * NROWS;
      v8h b0 = *(const v8h*)(bp);
      v8h b1 = *(const v8h*)(bp + 16);
      v16h bf;
      #pragma unroll
      for (int e = 0; e < 8; ++e) { bf[e] = b0[e]; bf[e + 8] = b1[e]; }
      acc[t] = __builtin_amdgcn_wmma_f32_16x16x32_f16(
          /*neg_a=*/false, af, /*neg_b=*/false, bf,
          /*c_mod=*/(short)0, acc[t], /*reuse_a=*/false, /*reuse_b=*/false);
    }
  }

  // Combine the two lane-halves: row Mrow's Z-sum lives in lanes Mrow, Mrow+16.
  float mtot = msum + __shfl_xor(msum, 16, 32);
  const float invN = 1.0f / (float)NROWS;
  float mv[8];
  #pragma unroll
  for (int v = 0; v < 8; ++v)
    mv[v] = __shfl(mtot, v + 8 * khalf, 32) * invN;  // m for C/D row v+8*khalf

  // Epilogue: C/D layout — lane -> col = t*16 + (lane&15), vgpr v -> row.
  #pragma unroll
  for (int t = 0; t < 16; ++t) {
    const int c = t * 16 + Mrow;
    const float bval = br[c];
    #pragma unroll
    for (int v = 0; v < 8; ++v) {
      const int i = i0 + v + 8 * khalf;
      out[(size_t)i * CDIM + c] =
          (r[(size_t)i * CDIM + c] + bval) * mv[v] + acc[t][v] * invN;
    }
  }
}

// ---------------------------------------------------------------------------
extern "C" void kernel_launch(void* const* d_in, const int* in_sizes, int n_in,
                              void* d_out, int out_size, void* d_ws, size_t ws_size,
                              hipStream_t stream) {
  const float* x  = (const float*)d_in[0];  // [N, C]
  const float* wr = (const float*)d_in[1];  // [C, C]
  const float* br = (const float*)d_in[2];  // [C]
  const float* wc = (const float*)d_in[3];  // [1, C]
  const float* bc = (const float*)d_in[4];  // [1]
  float* out = (float*)d_out;

  // Workspace layout (all 16B-aligned):
  //   r   : N*C f32   =  8 MiB   @ 0
  //   rhT : C*N f16   =  4 MiB   @ 8 MiB
  //   wrt : C*C f32   = 256 KiB  @ 12 MiB
  //   s   : (N+1) f32 =  32 KiB  @ 12.25 MiB
  char* ws = (char*)d_ws;
  float*    r   = (float*)ws;
  _Float16* rhT = (_Float16*)(ws + (size_t)NROWS * CDIM * 4);
  float*    wrt = (float*)(ws + (size_t)NROWS * CDIM * 6);
  float*    s   = (float*)(ws + (size_t)NROWS * CDIM * 6 + (size_t)CDIM * CDIM * 4);

  kT   <<<CDIM,       CDIM, 0, stream>>>(wr, wrt);
  kA   <<<NROWS / 4,  CDIM, 0, stream>>>(x, wrt, r, rhT);
  kB   <<<NROWS + 1,  CDIM, 0, stream>>>(r, wc, br, bc, s);
  kMain<<<NROWS / 64, 128,  0, stream>>>(s, rhT, r, br, out);
}
